// BilateralLayerTorch_34419867910150
// MI455X (gfx1250) — matrix operand order repaired and checked by reading, not compiled, not run on hardware
//
#include <hip/hip_runtime.h>
#include <hip/hip_bf16.h>

// Bilateral layer: splat -> 3x3x3 conv (WMMA f16->f32) -> trilinear slice.
// Shapes: input (2,16,1024,1024) f32, guide (2,1024,1024) f32,
//         conv_w (16,16,3,3,3) f32, conv_b (16) f32, out (2,16,1024,1024) f32.
// Bilateral grid: (b, u=128, v=128, s=8) cells.
// Layouts chosen for vectorized access:
//   grid_h : (b,u,v,s, ci) f16  -> conv A-fragment loads are 16B b128 loads
//   grid_c : (b,u,v,s, co) f32  -> slice corner loads are 64B (4x b128)

typedef __attribute__((ext_vector_type(16))) _Float16 v16h;
typedef __attribute__((ext_vector_type(8)))  _Float16 v8h;
typedef __attribute__((ext_vector_type(8)))  float    v8f;
typedef __attribute__((ext_vector_type(16))) float    v16f;

#define GH 128
#define GW 128
#define SR 8
#define CI 16
#define CO 16
#define BS 2
#define NCELL (GH * GW * SR)          // 131072 cells per batch
#define NKC 14                        // K = 27 taps * 16 ci = 432 -> 448 = 14*32

// ---------------------------------------------------------------------------
// Kernel 0: pack conv weights into WMMA B-fragment lane order (f16).
// B is (K=32) x (N=16) per step kc; lane = (hi = lane>>4, n = lane&15),
// element e in v16h holds B[klocal = (e < 8 ? hi*8 + e : 16 + hi*8 + (e-8))][n].
// kglobal = kc*32 + klocal ; tap = kglobal>>4 ; ci = kglobal&15 ; tap 27 = pad.
// ---------------------------------------------------------------------------
__global__ void pack_wfrag_kernel(const float* __restrict__ conv_w,
                                  _Float16* __restrict__ wfrag) {
    const int id = blockIdx.x * blockDim.x + threadIdx.x;
    if (id >= NKC * 32 * 16) return;
    const int e    = id & 15;
    const int lane = (id >> 4) & 31;
    const int kc   = id >> 9;
    const int hi   = lane >> 4;
    const int n    = lane & 15;               // co
    const int klocal = (e < 8) ? (hi * 8 + e) : (16 + hi * 8 + (e - 8));
    const int kg     = kc * 32 + klocal;
    const int tap    = kg >> 4;               // 0..27 (27 = zero pad)
    const int ci     = kg & 15;
    float v = 0.0f;
    if (tap < 27) v = conv_w[(n * CI + ci) * 27 + tap]; // (co,ci,kd,kh,kw) flat
    wfrag[id] = (_Float16)v;
}

// ---------------------------------------------------------------------------
// Kernel 1: splat. One block = one (b, u0, v0) grid cell column = 8x8 pixels.
// Pixels of a tile map ONLY to this cell -> LDS-only accumulation (ds_add_f32),
// no global atomics. 64 threads; coalesced ci-innermost f16 store.
// ---------------------------------------------------------------------------
__global__ void splat_kernel(const float* __restrict__ input,
                             const float* __restrict__ guide,
                             _Float16* __restrict__ grid_h) {
    __shared__ float cell[CI * SR]; // cell[ci*8 + s]
    const int t   = threadIdx.x;    // 0..63
    const int blk = blockIdx.x;     // BS*128*128
    const int v0  = blk & 127;
    const int u0  = (blk >> 7) & 127;
    const int b   = blk >> 14;

    cell[t] = 0.0f;
    cell[t + 64] = 0.0f;
    __syncthreads();

    const int py = u0 * 8 + (t >> 3);
    const int px = v0 * 8 + (t & 7);

    const float g    = guide[((long)b * 1024 + py) * 1024 + px];
    const float gpos = g * (float)SR;
    float lbf = floorf(gpos - 0.5f);
    if (lbf < 0.0f) lbf = 0.0f;
    const int   lb  = (int)lbf;
    const int   ub  = min(lb + 1, SR - 1);
    const float wgt = fabsf(gpos - 0.5f - lbf);
    const float nrm = 1.0f / 64.0f;
    const float w0  = (1.0f - wgt) * nrm;
    const float w1  = wgt * nrm;

    const long base = ((long)(b * CI) * 1024 + py) * 1024 + px;
    #pragma unroll 4
    for (int ci = 0; ci < CI; ++ci) {
        const float x = input[base + (long)ci * (1024 * 1024)];
        atomicAdd(&cell[ci * SR + lb], w0 * x);   // ds_add_f32
        atomicAdd(&cell[ci * SR + ub], w1 * x);
    }
    __syncthreads();

    // grid_h[(b,u0,v0,s,ci)] ; i = s*16 + ci -> contiguous coalesced stores
    const long obase = (((long)(b * GH + u0) * GW + v0) * SR) * CI;
    #pragma unroll
    for (int i = t; i < SR * CI; i += 64) {
        const int s = i >> 4, ci = i & 15;
        grid_h[obase + i] = (_Float16)cell[ci * SR + s];
    }
}

// ---------------------------------------------------------------------------
// Kernel 2: 3D conv as WMMA GEMM. One wave (32 threads) per output tile:
//   M = 16 spatial positions (2 v-rows x 8 s), N = 16 co, K = 448 (14 x 32).
// A-fragment (16-bit A 16x32 layout): lane m = lane&15 is the spatial row;
//   elements 0..7  -> tap 2*kc,   ci = hi*8 + e   (one 16B b128 load)
//   elements 8..15 -> tap 2*kc+1, ci = hi*8 + e-8 (one 16B b128 load)
// B-fragment: one 32B vector load from pre-packed wfrag.
// Tap offsets are compile-time constants (full unroll) -> cheap bounds checks.
// ---------------------------------------------------------------------------
__global__ void conv_wmma_kernel(const _Float16* __restrict__ grid_h,
                                 const _Float16* __restrict__ wfrag,
                                 const float* __restrict__ conv_b,
                                 float* __restrict__ grid_c) {
    const int lane = threadIdx.x;     // 0..31, one full wave, EXEC all-1s
    const int tile = blockIdx.x;      // BS * 128 * 64
    const int vp   = tile & 63;
    const int u    = (tile >> 6) & 127;
    const int b    = tile >> 13;
    const int v0   = vp * 2;

    const int hi = lane >> 4;
    const int m  = lane & 15;         // A spatial row
    const int av = v0 + (m >> 3);
    const int as = m & 7;

    v8f c = {0.f, 0.f, 0.f, 0.f, 0.f, 0.f, 0.f, 0.f};

    #pragma unroll
    for (int kc = 0; kc < NKC; ++kc) {
        v16h a;
        #pragma unroll
        for (int half = 0; half < 2; ++half) {
            const int tap = 2 * kc + half;
            v8h frag = {};
            if (tap < 27) {
                const int od = tap / 9 - 1;
                const int oh = (tap / 3) % 3 - 1;
                const int ow = tap % 3 - 1;
                const int uu = u + od, vv = av + oh, ss = as + ow;
                const bool ok = ((unsigned)uu < (unsigned)GH) &&
                                ((unsigned)vv < (unsigned)GW) &&
                                ((unsigned)ss < (unsigned)SR);
                if (ok) {
                    const long cellb =
                        (((long)(b * GH + uu) * GW + vv) * SR + ss) * CI;
                    frag = *(const v8h*)(grid_h + cellb + hi * 8); // 16B, aligned
                }
            }
            #pragma unroll
            for (int e = 0; e < 8; ++e) a[half * 8 + e] = frag[e];
        }
        const v16h bm = *(const v16h*)(wfrag + (((long)kc * 32 + lane) << 4));
        c = __builtin_amdgcn_wmma_f32_16x16x32_f16(
                /*neg_a=*/false, a, /*neg_b=*/false, bm,
                /*c_mod=*/(short)0, c, /*reuse_a=*/false, /*reuse_b=*/false);
    }

    // D layout: lane n = lane&15 (co), VGPR r holds M row = hi*8 + r.
    // Store co-innermost: lanes 0..15 / 16..31 each cover a 16-float run.
    const int   n    = lane & 15;
    const float bias = conv_b[n];
    #pragma unroll
    for (int r = 0; r < 8; ++r) {
        const int mm  = hi * 8 + r;
        const int dv  = v0 + (mm >> 3);
        const int ds_ = mm & 7;
        grid_c[(((long)(b * GH + u) * GW + dv) * SR + ds_) * CO + n] =
            c[r] + bias;
    }
}

// ---------------------------------------------------------------------------
// Kernel 3: slice. One thread per pixel (b,y,x); computes the 8 trilinear
// weights once, loads each corner as 16 contiguous co floats (64B vector),
// blends with vector FMAs, stores 16 channel planes (coalesced in x).
// Grid (16 MB f32) is served from L2.
// ---------------------------------------------------------------------------
__global__ void slice_kernel(const float* __restrict__ grid_c,
                             const float* __restrict__ guide,
                             float* __restrict__ out) {
    const long id = (long)blockIdx.x * blockDim.x + threadIdx.x; // BS*1024*1024
    const int x = (int)(id & 1023);
    const int y = (int)((id >> 10) & 1023);
    const int b = (int)(id >> 20);

    const float gx  = ((float)x + 0.5f) * 0.125f;   // /w * gw_ = /8
    const float gy  = ((float)y + 0.5f) * 0.125f;
    const float fxf = floorf(gx - 0.5f);
    const float fyf = floorf(gy - 0.5f);
    const float wx  = gx - 0.5f - fxf;
    const float wy  = gy - 0.5f - fyf;
    const int fx  = (int)fxf;
    const int fy  = (int)fyf;
    const int cx  = min(fx + 1, GW - 1);
    const int cy  = min(fy + 1, GH - 1);
    const int fxi = (fx < 0) ? fx + GW : fx;        // reference wraps negatives
    const int fyi = (fy < 0) ? fy + GH : fy;

    const float g  = guide[id];
    const float gz = g * (float)SR;
    float fzf = floorf(gz - 0.5f);
    if (fzf < 0.0f) fzf = 0.0f;
    const float wz = fabsf(gz - 0.5f - fzf);
    const int   fz = (int)fzf;
    const int   cz = min(fz + 1, SR - 1);

    const float w_ff = (1.0f - wx) * (1.0f - wy);
    const float w_fc = wx * (1.0f - wy);
    const float w_cf = (1.0f - wx) * wy;
    const float w_cc = wx * wy;
    const float wz0 = 1.0f - wz, wz1 = wz;

    const float* gb = grid_c + (long)b * NCELL * CO;
    const long c_ff = ((long)(fyi * GW + fxi) * SR);
    const long c_fc = ((long)(fyi * GW + cx ) * SR);
    const long c_cf = ((long)(cy  * GW + fxi) * SR);
    const long c_cc = ((long)(cy  * GW + cx ) * SR);

    v16f acc;
    {   // (fy,fx)
        const v16f lf = *(const v16f*)(gb + (c_ff + fz) * CO);
        const v16f lc = *(const v16f*)(gb + (c_ff + cz) * CO);
        acc = lf * (w_ff * wz0) + lc * (w_ff * wz1);
    }
    {   // (cy,fx)
        const v16f lf = *(const v16f*)(gb + (c_cf + fz) * CO);
        const v16f lc = *(const v16f*)(gb + (c_cf + cz) * CO);
        acc += lf * (w_cf * wz0) + lc * (w_cf * wz1);
    }
    {   // (fy,cx)
        const v16f lf = *(const v16f*)(gb + (c_fc + fz) * CO);
        const v16f lc = *(const v16f*)(gb + (c_fc + cz) * CO);
        acc += lf * (w_fc * wz0) + lc * (w_fc * wz1);
    }
    {   // (cy,cx)
        const v16f lf = *(const v16f*)(gb + (c_cc + fz) * CO);
        const v16f lc = *(const v16f*)(gb + (c_cc + cz) * CO);
        acc += lf * (w_cc * wz0) + lc * (w_cc * wz1);
    }

    const long obase = (long)b * CO * (1024 * 1024) + (long)y * 1024 + x;
    #pragma unroll
    for (int co = 0; co < CO; ++co)
        out[obase + (long)co * (1024 * 1024)] = acc[co];
}

// ---------------------------------------------------------------------------
// Launch. Workspace: [wfrag f16 16KB][grid_h f16 8MB][grid_c f32 16MB] ~ 25MB.
// ---------------------------------------------------------------------------
extern "C" void kernel_launch(void* const* d_in, const int* in_sizes, int n_in,
                              void* d_out, int out_size, void* d_ws, size_t ws_size,
                              hipStream_t stream) {
    const float* input  = (const float*)d_in[0];
    const float* guide  = (const float*)d_in[1];
    const float* conv_w = (const float*)d_in[2];
    const float* conv_b = (const float*)d_in[3];
    float*       out    = (float*)d_out;

    char* ws = (char*)d_ws;
    _Float16* wfrag  = (_Float16*)ws;                                 // 14,336 B
    _Float16* grid_h = (_Float16*)(ws + 16384);                       // 8,388,608 B
    float*    grid_c = (float*)(ws + 16384 + (size_t)BS * NCELL * CI * 2);

    pack_wfrag_kernel<<<(NKC * 32 * 16 + 255) / 256, 256, 0, stream>>>(conv_w, wfrag);
    splat_kernel<<<BS * GH * GW, 64, 0, stream>>>(input, guide, grid_h);
    conv_wmma_kernel<<<BS * GH * (GW / 2), 32, 0, stream>>>(grid_h, wfrag, conv_b, grid_c);
    slice_kernel<<<(BS * 1024 * 1024) / 256, 256, 0, stream>>>(grid_c, guide, out);
}
